// MySelfAttn_2482491097219
// MI455X (gfx1250) — matrix operand order repaired and checked by reading, not compiled, not run on hardware
//
#include <hip/hip_runtime.h>
#include <hip/hip_bf16.h>
#include <cstdint>
#include <cstddef>

// Problem constants (match reference)
#define B_ 32
#define S_ 512
#define E_ 1024
#define H_ 8
#define D_ 128
#define F_ 8192

// Trivial vector types (safe in unions)
typedef __attribute__((ext_vector_type(16))) __bf16  bf16x16;
typedef __attribute__((ext_vector_type(8)))  float   f32x8;
typedef __attribute__((ext_vector_type(4)))  float   f32x4;
typedef __attribute__((ext_vector_type(4)))  unsigned int u32x4;
typedef __attribute__((ext_vector_type(8)))  unsigned int u32x8;
typedef __attribute__((ext_vector_type(2)))  unsigned int u32x2;

union Frag16 { bf16x16 v; u32x4 q[2]; };   // 32 bytes: one WMMA A/B operand

__device__ __forceinline__ unsigned short f32_to_bf16(float f) {
  union { float f; unsigned int u; } c; c.f = f;
  unsigned int u = c.u;
  u += 0x7FFFu + ((u >> 16) & 1u);         // round-to-nearest-even
  return (unsigned short)(u >> 16);
}

__device__ __forceinline__ f32x8 zero8() {
  f32x8 z;
#pragma unroll
  for (int i = 0; i < 8; ++i) z[i] = 0.0f;
  return z;
}

// ---- CDNA5 async Global->LDS copy (ASYNCcnt), GVS addressing ----
__device__ __forceinline__ void async_b128(unsigned lds_off, const void* base, int gbyte) {
  asm volatile("global_load_async_to_lds_b128 %0, %1, %2"
               :: "v"(lds_off), "v"(gbyte), "s"(base) : "memory");
}
__device__ __forceinline__ void wait_async0() {
  asm volatile("s_wait_asynccnt 0x0" ::: "memory");
}
__device__ __forceinline__ void wait_async4() {   // one 4-op stage still in flight
  asm volatile("s_wait_asynccnt 0x4" ::: "memory");
}
__device__ __forceinline__ unsigned lds_addr_of(const void* p) {
  return (unsigned)(uintptr_t)p;      // flat LDS aperture: low 32 bits = LDS offset
}

// ---- CDNA5 Tensor Data Mover: 2D tile Global->LDS (TENSORcnt) ----
// 2-group D# form (VADDR2/3 = NULL): group0 = 4 SGPRs, group1 = 8 SGPRs.
__device__ __forceinline__ void tdm_load_2d(unsigned lds_off, const void* gaddr,
                                            unsigned tensor_d0, unsigned tensor_d1,
                                            unsigned tile_d0, unsigned tile_d1,
                                            unsigned stride_elems) {
  const unsigned long long ga = (unsigned long long)(uintptr_t)gaddr;
  u32x4 g0;
  g0[0] = 1u;                                        // count=1, user-mode D#
  g0[1] = lds_off;                                   // lds_addr
  g0[2] = (unsigned)(ga & 0xFFFFFFFFu);              // global_addr[31:0]
  g0[3] = (unsigned)((ga >> 32) & 0x01FFFFFFu) | (2u << 30);  // addr[56:32]|type=2
  u32x8 g1;
  g1[0] = (1u << 16);                                // data_size=1 (2-byte elems)
  g1[1] = (tensor_d0 & 0xFFFFu) << 16;               // tensor_dim0[15:0] @bit48
  g1[2] = ((tensor_d0 >> 16) & 0xFFFFu) | ((tensor_d1 & 0xFFFFu) << 16);
  g1[3] = ((tensor_d1 >> 16) & 0xFFFFu) | (tile_d0 << 16);  // tile_dim0 @bit112
  g1[4] = tile_d1;                                   // tile_dim1; tile_dim2=0
  g1[5] = stride_elems;                              // tensor_dim0_stride[31:0]
  g1[6] = 0;                                         // stride hi / dim1_stride lo
  g1[7] = 0;
  asm volatile("tensor_load_to_lds %0, %1" :: "s"(g0), "s"(g1) : "memory");
}

// ---------------------------------------------------------------------------
// One-time f32 -> bf16 conversion (x), 4 elements/thread.
// ---------------------------------------------------------------------------
__global__ __launch_bounds__(256)
void to_bf16(const float* __restrict__ src, unsigned short* __restrict__ dst) {
  const size_t i = ((size_t)blockIdx.x * 256 + threadIdx.x) * 4;
  f32x4 f = *(const f32x4*)(src + i);
  u32x2 p;
  p.x = (unsigned)f32_to_bf16(f.x) | ((unsigned)f32_to_bf16(f.y) << 16);
  p.y = (unsigned)f32_to_bf16(f.z) | ((unsigned)f32_to_bf16(f.w) << 16);
  *(u32x2*)(dst + i) = p;
}

// ---------------------------------------------------------------------------
// One-time weight prep: W[K,N] f32 -> Wt[N,K] bf16 (tiled transpose via LDS)
// ---------------------------------------------------------------------------
__global__ __launch_bounds__(256)
void transpose_to_bf16(const float* __restrict__ W, unsigned short* __restrict__ Wt,
                       int K, int N) {
  __shared__ unsigned short t[32][33];
  const int tx = threadIdx.x, ty = threadIdx.y;
  const int n0 = blockIdx.x * 32, k0 = blockIdx.y * 32;
#pragma unroll
  for (int i = 0; i < 32; i += 8)
    t[ty + i][tx] = f32_to_bf16(W[(size_t)(k0 + ty + i) * N + n0 + tx]);
  __syncthreads();
#pragma unroll
  for (int i = 0; i < 32; i += 8)
    Wt[(size_t)(n0 + ty + i) * K + k0 + tx] = t[tx][ty + i];
}

// ---------------------------------------------------------------------------
// Tiled GEMM:  Out = epilogue(A[M,K](bf16) * Bt[N,K](bf16)^T + bias)
// Block tile 128x128, BK=64, 256 threads = 8 waves; wave = 32x64 sub-tile.
// A tile: async Global->LDS (all threads).  B tile: TDM DMA (wave 0 only).
// Both double-buffered (ASYNCcnt<=4 / TENSORcnt<=1).
// ---------------------------------------------------------------------------
enum { EPI_QK_HEADED = 0, EPI_V_TRANS = 1, EPI_GELU_BF16 = 2, EPI_OUT_F32 = 3 };

template <int EPI>
__global__ __launch_bounds__(256)
void gemm_wmma(const unsigned short* __restrict__ A,
               const unsigned short* __restrict__ Bt,
               const float* __restrict__ bias, void* __restrict__ Out,
               int M, int N, int K) {
  __shared__ __attribute__((aligned(16))) unsigned short sA[2][128 * 64]; // [m][k]
  __shared__ __attribute__((aligned(16))) unsigned short sB[2][128 * 64]; // [n][k]

  const int tid   = threadIdx.x;
  const int wave  = tid >> 5;
  const int lane  = tid & 31;
  const int l16   = lane & 15;
  const int lhalf = lane >> 4;
  const int bn = blockIdx.x;      // N / 128
  const int bm = blockIdx.y;      // M / 128
  const int wm = wave & 3;        // 4 waves along M (32 rows each)
  const int wn = wave >> 2;       // 2 waves along N (64 cols each)

  const unsigned sA_lds = lds_addr_of(&sA[0][0]);
  const unsigned sB_lds = lds_addr_of(&sB[0][0]);
  const int crow = tid >> 1;               // 0..127 (A copy row)
  const int ccol = (tid & 1) * 32;         // 0 or 32

  auto issue_stage = [&](int s, int buf) {
    // A tile: 4 async b128 per thread
    const int ga = (int)(((size_t)(bm * 128 + crow) * (size_t)K + (size_t)s * 64 + ccol) * 2);
    const unsigned la = sA_lds + (unsigned)(buf * (128 * 64 * 2) + (crow * 64 + ccol) * 2);
#pragma unroll
    for (int j = 0; j < 4; ++j) async_b128(la + j * 16, A, ga + j * 16);
    // B tile: one TDM descriptor (wave 0), 128 rows x 64 elems, row stride K
    if (tid < 32) {
      const unsigned short* gtile = Bt + (size_t)(bn * 128) * (size_t)K + (size_t)s * 64;
      tdm_load_2d(sB_lds + (unsigned)(buf * (128 * 64 * 2)), gtile,
                  (unsigned)K, (unsigned)N, 64u, 128u, (unsigned)K);
    }
  };

  f32x8 acc[2][4];
#pragma unroll
  for (int r = 0; r < 2; ++r)
#pragma unroll
    for (int c = 0; c < 4; ++c) acc[r][c] = zero8();

  const int ns = K >> 6;
  issue_stage(0, 0);
  for (int s = 0; s < ns; ++s) {
    const int buf = s & 1;
    if (s + 1 < ns) {
      issue_stage(s + 1, buf ^ 1);
      wait_async4();
      if (tid < 32) __builtin_amdgcn_s_wait_tensorcnt(1);
    } else {
      wait_async0();
      if (tid < 32) __builtin_amdgcn_s_wait_tensorcnt(0);
    }
    __syncthreads();

#pragma unroll
    for (int kc = 0; kc < 2; ++kc) {
      Frag16 a[2], b[4];
#pragma unroll
      for (int r = 0; r < 2; ++r) {
        const int row = wm * 32 + r * 16 + l16;
        const int kof = kc * 32 + lhalf * 8;       // A layout: lane<16 -> K{0-7,16-23}
        a[r].q[0] = *(const u32x4*)(&sA[buf][row * 64 + kof]);
        a[r].q[1] = *(const u32x4*)(&sA[buf][row * 64 + kof + 16]);
      }
#pragma unroll
      for (int c = 0; c < 4; ++c) {
        const int n   = wn * 64 + c * 16 + l16;
        const int kof = kc * 32 + lhalf * 8;       // B layout: lane<16 -> K{0-7,16-23}
        b[c].q[0] = *(const u32x4*)(&sB[buf][n * 64 + kof]);
        b[c].q[1] = *(const u32x4*)(&sB[buf][n * 64 + kof + 16]);
      }
#pragma unroll
      for (int r = 0; r < 2; ++r)
#pragma unroll
        for (int c = 0; c < 4; ++c)
          acc[r][c] = __builtin_amdgcn_wmma_f32_16x16x32_bf16(
              false, a[r].v, false, b[c].v, (short)0, acc[r][c], false, false);
    }
    __syncthreads();   // all waves done with buf before DMA overwrites it
  }

  // ---- epilogue ----
#pragma unroll
  for (int r = 0; r < 2; ++r) {
#pragma unroll
    for (int c = 0; c < 4; ++c) {
#pragma unroll
      for (int i = 0; i < 8; ++i) {
        const int row = bm * 128 + wm * 32 + r * 16 + i + lhalf * 8; // C layout
        const int col = bn * 128 + wn * 64 + c * 16 + l16;
        float v = acc[r][c][i] + bias[col];
        if (EPI == EPI_QK_HEADED) {           // -> bf16 [B,H,S,D]
          const int bb = row >> 9, ss = row & (S_ - 1);
          const int hh = col >> 7, dd = col & (D_ - 1);
          ((unsigned short*)Out)[(((size_t)(bb * H_ + hh)) * S_ + ss) * D_ + dd] =
              f32_to_bf16(v);
        } else if (EPI == EPI_V_TRANS) {      // -> bf16 [B,H,D,S]
          const int bb = row >> 9, ss = row & (S_ - 1);
          const int hh = col >> 7, dd = col & (D_ - 1);
          ((unsigned short*)Out)[(((size_t)(bb * H_ + hh)) * D_ + dd) * S_ + ss] =
              f32_to_bf16(v);
        } else if (EPI == EPI_GELU_BF16) {    // exact-erf GELU -> bf16 row-major
          float g = 0.5f * v * (1.0f + erff(v * 0.70710678118654752440f));
          ((unsigned short*)Out)[(size_t)row * (size_t)N + col] = f32_to_bf16(g);
        } else {                              // f32 row-major (final output)
          ((float*)Out)[(size_t)row * (size_t)N + col] = v;
        }
      }
    }
  }
}

// ---------------------------------------------------------------------------
// Flash-style attention (unscaled scores, per reference).
// Block = 8 waves (one 16-row q tile each); grid = (B*H, S/128).
// K/V tiles async-staged to LDS (double-buffered), shared by all waves.
// Output written as bf16 [B,S,E] (feeds FFN1 directly).
// ---------------------------------------------------------------------------
__global__ __launch_bounds__(256)
void attn_wmma(const unsigned short* __restrict__ Q,   // bf16 [B,H,S,D]
               const unsigned short* __restrict__ Kk,  // bf16 [B,H,S,D]
               const unsigned short* __restrict__ Vt,  // bf16 [B,H,D,S]
               unsigned short* __restrict__ Ob) {      // bf16 [B,S,E]
  __shared__ __attribute__((aligned(16))) unsigned short sK[2][32 * 128];  // [key][d]
  __shared__ __attribute__((aligned(16))) unsigned short sV[2][128 * 32];  // [d][key]
  __shared__ __attribute__((aligned(16))) unsigned short sP[8][16 * 32];   // per-wave P

  const int tid   = threadIdx.x;
  const int wave  = tid >> 5;
  const int lane  = tid & 31;
  const int l16   = lane & 15;
  const int lhalf = lane >> 4;
  const int bh = blockIdx.x;                  // b*H + h
  const int qb = blockIdx.y;                  // 128-row q block
  const int bb = bh / H_, hh = bh % H_;
  const int q0 = qb * 128 + wave * 16;

  const unsigned short* qp = Q  + (size_t)bh * S_ * D_;
  const unsigned short* kp = Kk + (size_t)bh * S_ * D_;
  const unsigned short* vp = Vt + (size_t)bh * D_ * S_;
  const unsigned sK_lds = lds_addr_of(&sK[0][0]);
  const unsigned sV_lds = lds_addr_of(&sV[0][0]);

  auto issue_kv = [&](int j, int buf) {                // j = key-block base
    const int key = tid >> 3, c0 = (tid & 7) * 16;
    const int gk = (int)(((size_t)(j + key) * D_ + c0) * 2);
    const unsigned lk = sK_lds + (unsigned)(buf * (32 * 128 * 2) + (key * 128 + c0) * 2);
    async_b128(lk, kp, gk);
    async_b128(lk + 16, kp, gk + 16);
    const int dd = tid >> 1, c1 = (tid & 1) * 16;
    const int gv = (int)(((size_t)dd * S_ + j + c1) * 2);
    const unsigned lv = sV_lds + (unsigned)(buf * (128 * 32 * 2) + (dd * 32 + c1) * 2);
    async_b128(lv, vp, gv);
    async_b128(lv + 16, vp, gv + 16);
  };

  // q fragments: 4 chunks of 16x32 covering D=128 (loaded once)
  Frag16 qf[4];
#pragma unroll
  for (int kc = 0; kc < 4; ++kc) {
    const int row = q0 + l16;
    const int kof = kc * 32 + lhalf * 8;
    qf[kc].q[0] = *(const u32x4*)(qp + (size_t)row * D_ + kof);
    qf[kc].q[1] = *(const u32x4*)(qp + (size_t)row * D_ + kof + 16);
  }

  f32x8 o[8];
#pragma unroll
  for (int t = 0; t < 8; ++t) o[t] = zero8();
  float mrow[8], lrow[8];
#pragma unroll
  for (int i = 0; i < 8; ++i) { mrow[i] = -3.0e38f; lrow[i] = 0.0f; }

  issue_kv(0, 0);
  for (int it = 0; it < S_ / 32; ++it) {
    const int buf = it & 1;
    if (it + 1 < S_ / 32) { issue_kv((it + 1) * 32, buf ^ 1); wait_async4(); }
    else                  { wait_async0(); }
    __syncthreads();

    // ---- scores: two 16x16 tiles over this 32-key block ----
    f32x8 sc[2];
#pragma unroll
    for (int t = 0; t < 2; ++t) {
      f32x8 s = zero8();
#pragma unroll
      for (int kc = 0; kc < 4; ++kc) {
        Frag16 bf;
        const int key = t * 16 + l16;
        const int dof = kc * 32 + lhalf * 8;
        bf.q[0] = *(const u32x4*)(&sK[buf][key * 128 + dof]);
        bf.q[1] = *(const u32x4*)(&sK[buf][key * 128 + dof + 16]);
        s = __builtin_amdgcn_wmma_f32_16x16x32_bf16(
            false, qf[kc].v, false, bf.v, (short)0, s, false, false);
      }
      sc[t] = s;
    }

    // ---- online softmax (rows live in C-layout: VGPR i + lane half) ----
#pragma unroll
    for (int i = 0; i < 8; ++i) {
      float mx = fmaxf(sc[0][i], sc[1][i]);
#pragma unroll
      for (int msk = 8; msk >= 1; msk >>= 1)       // stays within 16-lane half
        mx = fmaxf(mx, __shfl_xor(mx, msk, 32));
      const float mn = fmaxf(mrow[i], mx);
      const float scale = __expf(mrow[i] - mn);
      mrow[i] = mn;
      const float p0 = __expf(sc[0][i] - mn);
      const float p1 = __expf(sc[1][i] - mn);
      sc[0][i] = p0; sc[1][i] = p1;
      float rs = p0 + p1;
#pragma unroll
      for (int msk = 8; msk >= 1; msk >>= 1)
        rs += __shfl_xor(rs, msk, 32);
      lrow[i] = lrow[i] * scale + rs;
#pragma unroll
      for (int t2 = 0; t2 < 8; ++t2) o[t2][i] *= scale;
    }

    // ---- transpose P (C layout -> A layout) through per-wave LDS ----
#pragma unroll
    for (int t = 0; t < 2; ++t)
#pragma unroll
      for (int i = 0; i < 8; ++i)
        sP[wave][(i + lhalf * 8) * 32 + t * 16 + l16] = f32_to_bf16(sc[t][i]);
    asm volatile("s_wait_dscnt 0x0" ::: "memory");   // per-wave tile: dscnt suffices

    Frag16 pf;
    {
      const int kof = lhalf * 8;
      pf.q[0] = *(const u32x4*)(&sP[wave][l16 * 32 + kof]);
      pf.q[1] = *(const u32x4*)(&sP[wave][l16 * 32 + kof + 16]);
    }

    // ---- o += P(16x32) x V(32x128), 8 WMMAs over d-tiles ----
#pragma unroll
    for (int t2 = 0; t2 < 8; ++t2) {
      Frag16 vf;
      const int d = t2 * 16 + l16;
      const int c = lhalf * 8;
      vf.q[0] = *(const u32x4*)(&sV[buf][d * 32 + c]);
      vf.q[1] = *(const u32x4*)(&sV[buf][d * 32 + c + 16]);
      o[t2] = __builtin_amdgcn_wmma_f32_16x16x32_bf16(
          false, pf.v, false, vf.v, (short)0, o[t2], false, false);
    }
    __syncthreads();   // this buf consumed; next restage may overwrite it
  }

  // ---- epilogue: normalize, write concat-head bf16 [B,S,E] ----
#pragma unroll
  for (int t2 = 0; t2 < 8; ++t2) {
#pragma unroll
    for (int i = 0; i < 8; ++i) {
      const int row = q0 + i + lhalf * 8;            // s
      const int col = hh * D_ + t2 * 16 + l16;       // e
      Ob[((size_t)bb * S_ + row) * E_ + col] = f32_to_bf16(o[t2][i] / lrow[i]);
    }
  }
}

// ---------------------------------------------------------------------------
extern "C" void kernel_launch(void* const* d_in, const int* in_sizes, int n_in,
                              void* d_out, int out_size, void* d_ws, size_t ws_size,
                              hipStream_t stream) {
  (void)in_sizes; (void)n_in; (void)out_size; (void)ws_size;
  const float* x  = (const float*)d_in[0];
  const float* Wq = (const float*)d_in[1];
  const float* bq = (const float*)d_in[2];
  const float* Wk = (const float*)d_in[3];
  const float* bk = (const float*)d_in[4];
  const float* Wv = (const float*)d_in[5];
  const float* bv = (const float*)d_in[6];
  const float* W1 = (const float*)d_in[7];
  const float* b1 = (const float*)d_in[8];
  const float* W2 = (const float*)d_in[9];
  const float* b2 = (const float*)d_in[10];

  // Workspace layout (all bf16)
  char* ws = (char*)d_ws;
  const size_t nBSE = (size_t)B_ * S_ * E_;                       // 16,777,216
  unsigned short* xb   = (unsigned short*)ws; ws += nBSE * 2;
  unsigned short* Wqt  = (unsigned short*)ws; ws += (size_t)E_ * E_ * 2;
  unsigned short* Wkt  = (unsigned short*)ws; ws += (size_t)E_ * E_ * 2;
  unsigned short* Wvt  = (unsigned short*)ws; ws += (size_t)E_ * E_ * 2;
  unsigned short* W1t  = (unsigned short*)ws; ws += (size_t)E_ * F_ * 2;
  unsigned short* W2t  = (unsigned short*)ws; ws += (size_t)F_ * E_ * 2;
  unsigned short* qb16 = (unsigned short*)ws; ws += nBSE * 2;
  unsigned short* kb16 = (unsigned short*)ws; ws += nBSE * 2;
  unsigned short* vt16 = (unsigned short*)ws; ws += nBSE * 2;
  unsigned short* Ob   = (unsigned short*)ws; ws += nBSE * 2;
  unsigned short* hid  = (unsigned short*)ws;                     // 256 MiB

  const int M = B_ * S_;                              // 16384
  dim3 blk(256);
  dim3 tblk(32, 8);

  // One-time precision/layout prep
  to_bf16<<<dim3((unsigned)(nBSE / 1024)), blk, 0, stream>>>(x, xb);
  transpose_to_bf16<<<dim3(E_ / 32, E_ / 32), tblk, 0, stream>>>(Wq, Wqt, E_, E_);
  transpose_to_bf16<<<dim3(E_ / 32, E_ / 32), tblk, 0, stream>>>(Wk, Wkt, E_, E_);
  transpose_to_bf16<<<dim3(E_ / 32, E_ / 32), tblk, 0, stream>>>(Wv, Wvt, E_, E_);
  transpose_to_bf16<<<dim3(F_ / 32, E_ / 32), tblk, 0, stream>>>(W1, W1t, E_, F_);
  transpose_to_bf16<<<dim3(E_ / 32, F_ / 32), tblk, 0, stream>>>(W2, W2t, F_, E_);

  // QKV projections (headed layouts; V transposed)
  gemm_wmma<EPI_QK_HEADED><<<dim3(E_ / 128, M / 128), blk, 0, stream>>>(
      xb, Wqt, bq, qb16, M, E_, E_);
  gemm_wmma<EPI_QK_HEADED><<<dim3(E_ / 128, M / 128), blk, 0, stream>>>(
      xb, Wkt, bk, kb16, M, E_, E_);
  gemm_wmma<EPI_V_TRANS><<<dim3(E_ / 128, M / 128), blk, 0, stream>>>(
      xb, Wvt, bv, vt16, M, E_, E_);

  // Attention
  attn_wmma<<<dim3(B_ * H_, S_ / 128), blk, 0, stream>>>(qb16, kb16, vt16, Ob);

  // FFN
  gemm_wmma<EPI_GELU_BF16><<<dim3(F_ / 128, M / 128), blk, 0, stream>>>(
      Ob, W1t, b1, hid, M, F_, E_);
  gemm_wmma<EPI_OUT_F32><<<dim3(E_ / 128, M / 128), blk, 0, stream>>>(
      hid, W2t, b2, d_out, M, E_, F_);
}
